// LSTM_13469017440744
// MI455X (gfx1250) — matrix operand order repaired and checked by reading, not compiled, not run on hardware
//
#include <hip/hip_runtime.h>
#include <math.h>

// ---------------------------------------------------------------------------
// LSTM (T=512, B=64, I=H=1024) for gfx1250.
// Fused [x_t | h_{t-1}] @ [W_ih | W_hh]^T per step (K = 2048) with
// v_wmma_f32_16x16x32_bf16 (fp32 accum, fp32 cell state). One persistent
// kernel (64 WGs x 256 thr) with a device-scope atomic barrier per timestep.
//
// v3: async global->LDS staging (GLOBAL_LOAD_ASYNC_TO_LDS_B128) now uses the
// saddr (SGPR base + 32-bit VGPR offset) addressing mode: the per-lane
// offsets and LDS destinations are computed once before the t-loop, so the
// per-slab cost is SALU-only base updates instead of the v_add_nc_u64 chains
// v2 generated (which pushed the kernel past 256 VGPRs). Slab loop is
// unroll-2 so the double-buffer index is compile-time constant.
// ---------------------------------------------------------------------------

#define T_STEPS 512
#define BATCH   64
#define IN_DIM  1024
#define HID     1024
#define G4      (4 * HID)        // 4096 gate rows
#define KTOT    (IN_DIM + HID)   // 2048
#define NWG     64
#define WGT     256
#define SLAB_K  128
#define NSLAB   (KTOT / SLAB_K)  // 16
#define ROWP    136              // padded LDS row length (elements, 272 B)

typedef __attribute__((ext_vector_type(16))) __bf16 v16bf;
typedef __attribute__((ext_vector_type(8)))  float  v8f;

union FragBF { v16bf bf; uint4 q[2]; };

__device__ __forceinline__ unsigned short f2bf(float f) {
  __bf16 b = (__bf16)f;                       // RNE fptrunc
  return __builtin_bit_cast(unsigned short, b);
}
__device__ __forceinline__ float sigf(float x) {
  return 1.0f / (1.0f + __expf(-x));
}
// Low 32 bits of a generic pointer into the LDS aperture are the wave-relative
// LDS byte address (ISA 10.2 aperture rules) -> usable as async-load VDST.
__device__ __forceinline__ unsigned lds_off(const void* p) {
  return (unsigned)(uintptr_t)p;
}

// --------------------------- prep kernels ----------------------------------

__global__ void lstm_pack_w(const float* __restrict__ Wih,
                            const float* __restrict__ Whh,
                            unsigned short* __restrict__ Wc) {
  size_t i = (size_t)blockIdx.x * blockDim.x + threadIdx.x;   // < 4096*2048
  int n = (int)(i >> 11);
  int k = (int)(i & 2047);
  float v = (k < IN_DIM) ? Wih[(size_t)n * IN_DIM + k]
                         : Whh[(size_t)n * HID + (k - IN_DIM)];
  Wc[i] = f2bf(v);
}

__global__ void lstm_bias(const float* __restrict__ bih,
                          const float* __restrict__ bhh,
                          float* __restrict__ bs) {
  int i = blockIdx.x * blockDim.x + threadIdx.x;
  if (i < G4) bs[i] = bih[i] + bhh[i];
}

__global__ void lstm_x2bf(const float* __restrict__ x,
                          unsigned short* __restrict__ xbf) {
  size_t i = (size_t)blockIdx.x * blockDim.x + threadIdx.x;   // < T*B*I
  xbf[i] = f2bf(x[i]);
}

__global__ void lstm_init_state(const float* __restrict__ h0,
                                const float* __restrict__ c0,
                                unsigned short* __restrict__ hbf,
                                float* __restrict__ cbuf,
                                unsigned* __restrict__ bar) {
  int i = blockIdx.x * blockDim.x + threadIdx.x;              // < 64*1024
  hbf[i] = f2bf(h0[i]);                                       // ping buffer 0
  cbuf[i] = c0[i];
  if (i < 4) bar[i] = 0;                                      // barrier state
}

// --------------------------- main persistent kernel ------------------------

__global__ void __launch_bounds__(WGT, 1)
lstm_persistent(const unsigned short* __restrict__ xbf,   // [T][B][I] bf16
                const unsigned short* __restrict__ Wc,    // [4H][K] bf16
                const float* __restrict__ bsum,           // [4H]
                unsigned short* __restrict__ hbf,         // [2][B][H] bf16
                float* __restrict__ cbuf,                 // [B][H]
                float* __restrict__ out,                  // [T][B][H]
                unsigned* __restrict__ bar) {
  __shared__ unsigned short As[2][BATCH][ROWP];           // A slab, dbl-buf
  __shared__ unsigned short Bs[2][64][ROWP];              // B slab, dbl-buf
  __shared__ float xch[4][2][8][32];                      // i,f tile exchange

  const int w    = blockIdx.x;            // owns h columns [16w, 16w+16)
  const int wave = threadIdx.x >> 5;
  const int lane = threadIdx.x & 31;
  const int hi   = lane >> 4;
  const int nl   = lane & 15;
  const int mt   = wave & 3;              // M tile (batch rows 16*mt..)
  const int gp   = wave >> 2;             // 0: gates (i,f)  1: gates (g,o)
  const int g0   = 2 * gp, g1 = 2 * gp + 1;

  const size_t nrow0 = (size_t)(g0 * HID + w * 16 + nl);  // Wc row, frag 0
  const size_t nrow1 = (size_t)(g1 * HID + w * 16 + nl);  // Wc row, frag 1
  const float  bi0   = bsum[nrow0];
  const float  bi1   = bsum[nrow1];

  // Loop-invariant per-lane addresses for the async slab fill: 4 A-loads +
  // 4 B-loads per wave (16 B/lane, two rows per instruction). Global side is
  // a 32-bit byte offset against a uniform per-slab SGPR base (GVS mode).
  unsigned aoffs[4], boffs[4], ldsA0[4], ldsB0[4];
  #pragma unroll
  for (int i = 0; i < 4; ++i) {
    const int r = (wave * 4 + i) * 2 + hi;                     // 0..63
    aoffs[i] = (unsigned)((r * IN_DIM + nl * 8) * 2);          // row stride 2KB
    boffs[i] = (unsigned)((((r >> 4) * HID + w * 16 + (r & 15)) * KTOT +
                           nl * 8) * 2);                       // row stride 4KB
    ldsA0[i] = lds_off(&As[0][r][nl * 8]);
    ldsB0[i] = lds_off(&Bs[0][r][nl * 8]);
  }
  const unsigned strideA = (unsigned)sizeof(As[0]);            // buf 0 -> 1
  const unsigned strideB = (unsigned)sizeof(Bs[0]);

  auto issue_slab = [&](int buf, const unsigned short* sbaseA,
                        const unsigned short* sbaseB) {
    const unsigned dA = buf ? strideA : 0u;
    const unsigned dB = buf ? strideB : 0u;
    #pragma unroll
    for (int i = 0; i < 4; ++i)
      asm volatile("global_load_async_to_lds_b128 %0, %1, %2"
                   :: "v"(ldsA0[i] + dA), "v"(aoffs[i]), "s"(sbaseA)
                   : "memory");
    #pragma unroll
    for (int i = 0; i < 4; ++i)
      asm volatile("global_load_async_to_lds_b128 %0, %1, %2"
                   :: "v"(ldsB0[i] + dB), "v"(boffs[i]), "s"(sbaseB)
                   : "memory");
  };

  // One 32-K WMMA chunk from LDS (contiguous k per lane half, k = kk + 16*hi + s, same for A and B).
  auto do_chunk = [&](int buf, int kk, v8f& A0, v8f& A1) {
    FragBF a, b0, b1;
    const unsigned short* ap = &As[buf][mt * 16 + nl][kk + 16 * hi];
    a.q[0] = *(const uint4*)ap;
    a.q[1] = *(const uint4*)(ap + 8);
    const unsigned short* bp0 = &Bs[buf][g0 * 16 + nl][kk + 16 * hi];
    b0.q[0] = *(const uint4*)bp0;
    b0.q[1] = *(const uint4*)(bp0 + 8);
    const unsigned short* bp1 = &Bs[buf][g1 * 16 + nl][kk + 16 * hi];
    b1.q[0] = *(const uint4*)bp1;
    b1.q[1] = *(const uint4*)(bp1 + 8);
    A0 = __builtin_amdgcn_wmma_f32_16x16x32_bf16(false, a.bf, false, b0.bf,
                                                 (short)0, A0, false, false);
    A1 = __builtin_amdgcn_wmma_f32_16x16x32_bf16(false, a.bf, false, b1.bf,
                                                 (short)0, A1, false, false);
  };

  #pragma unroll 1
  for (int t = 0; t < T_STEPS; ++t) {
    const unsigned short* xbase = xbf + (size_t)t * BATCH * IN_DIM;
    const unsigned short* hbase = hbf + (size_t)(t & 1) * BATCH * HID;

    v8f acc0a = {}, acc0b = {}, acc1a = {}, acc1b = {};
    issue_slab(0, xbase, Wc);                     // prologue: slab 0 (x data)
    #pragma unroll 2
    for (int s = 0; s < NSLAB; ++s) {
      const int buf = s & 1;
      asm volatile("s_wait_asynccnt 0x0" ::: "memory");   // slab s landed
      __syncthreads();                                    // ... for all waves
      if (s + 1 < NSLAB) {                                // overlap next fill
        const int k0n = (s + 1) * SLAB_K;
        const unsigned short* an =
            (k0n < IN_DIM) ? (xbase + k0n) : (hbase + (k0n - IN_DIM));
        issue_slab(buf ^ 1, an, Wc + k0n);
      }
      do_chunk(buf,  0, acc0a, acc1a);            // 4 indep. WMMA chains
      do_chunk(buf, 32, acc0b, acc1b);
      do_chunk(buf, 64, acc0a, acc1a);
      do_chunk(buf, 96, acc0b, acc1b);
    }

    v8f acc0, acc1;
    #pragma unroll
    for (int r = 0; r < 8; ++r) {
      acc0[r] = acc0a[r] + acc0b[r] + bi0;
      acc1[r] = acc1a[r] + acc1b[r] + bi1;
    }

    // exchange (i,f) tiles from waves 0-3 to waves 4-7 via LDS
    __syncthreads();
    if (gp == 0) {
      #pragma unroll
      for (int r = 0; r < 8; ++r) {
        xch[mt][0][r][lane] = acc0[r];                    // i gate
        xch[mt][1][r][lane] = acc1[r];                    // f gate
      }
    }
    __syncthreads();
    if (gp == 1) {
      unsigned short* hnext = hbf + (size_t)((t + 1) & 1) * BATCH * HID;
      #pragma unroll
      for (int r = 0; r < 8; ++r) {
        float iv = sigf(xch[mt][0][r][lane]);
        float fv = sigf(xch[mt][1][r][lane]);
        float gv = tanhf(acc0[r]);                        // g gate
        float ov = sigf(acc1[r]);                         // o gate
        int    m   = mt * 16 + r + 8 * hi;                // C/D layout row
        size_t idx = (size_t)m * HID + (size_t)w * 16 + nl;
        float cv = fv * cbuf[idx] + iv * gv;
        cbuf[idx] = cv;
        float hv = ov * tanhf(cv);
        out[(size_t)t * BATCH * HID + idx] = hv;
        hnext[idx] = f2bf(hv);
      }
    }

    // grid-wide barrier: h(t) fully written before anyone reads it at t+1
    __threadfence();
    __syncthreads();
    if (threadIdx.x == 0) {
      unsigned g = __hip_atomic_load(bar + 1, __ATOMIC_ACQUIRE,
                                     __HIP_MEMORY_SCOPE_AGENT);
      unsigned a = __hip_atomic_fetch_add(bar, 1u, __ATOMIC_ACQ_REL,
                                          __HIP_MEMORY_SCOPE_AGENT);
      if (a == NWG - 1) {
        __hip_atomic_store(bar, 0u, __ATOMIC_RELAXED, __HIP_MEMORY_SCOPE_AGENT);
        __hip_atomic_fetch_add(bar + 1, 1u, __ATOMIC_RELEASE,
                               __HIP_MEMORY_SCOPE_AGENT);
      } else {
        while (__hip_atomic_load(bar + 1, __ATOMIC_ACQUIRE,
                                 __HIP_MEMORY_SCOPE_AGENT) == g) {
          __builtin_amdgcn_s_sleep(8);
        }
      }
    }
    __syncthreads();
  }
}

// ---------------------------------------------------------------------------

extern "C" void kernel_launch(void* const* d_in, const int* in_sizes, int n_in,
                              void* d_out, int out_size, void* d_ws,
                              size_t ws_size, hipStream_t stream) {
  const float* x   = (const float*)d_in[0];   // [T,B,I]
  const float* h0  = (const float*)d_in[1];   // [1,B,H]
  const float* c0  = (const float*)d_in[2];   // [1,B,H]
  const float* Wih = (const float*)d_in[3];   // [4H,I]
  const float* Whh = (const float*)d_in[4];   // [4H,H]
  const float* bih = (const float*)d_in[5];   // [4H]
  const float* bhh = (const float*)d_in[6];   // [4H]
  float* out = (float*)d_out;                 // [T,B,H]

  char* ws = (char*)d_ws;
  // 256B-aligned workspace carve-up (total 84,427,008 bytes)
  unsigned short* Wc  = (unsigned short*)(ws + 0);          // 16,777,216 B
  float*          bs  = (float*)(ws + 16777216);            //     16,384 B
  unsigned short* xbf = (unsigned short*)(ws + 16793600);   // 67,108,864 B
  unsigned short* hb  = (unsigned short*)(ws + 83902464);   //    262,144 B
  float*          cb  = (float*)(ws + 84164608);            //    262,144 B
  unsigned*       bar = (unsigned*)(ws + 84426752);         //        256 B

  lstm_pack_w<<<(G4 * KTOT) / WGT, WGT, 0, stream>>>(Wih, Whh, Wc);
  lstm_bias<<<(G4 + WGT - 1) / WGT, WGT, 0, stream>>>(bih, bhh, bs);
  lstm_x2bf<<<(T_STEPS * BATCH * IN_DIM) / WGT, WGT, 0, stream>>>(x, xbf);
  lstm_init_state<<<(BATCH * HID) / WGT, WGT, 0, stream>>>(h0, c0, hb, cb, bar);
  lstm_persistent<<<NWG, WGT, 0, stream>>>(xbf, Wc, bs, hb, cb, out, bar);
}
